// E2E_3444563771448
// MI455X (gfx1250) — matrix operand order repaired and checked by reading, not compiled, not run on hardware
//
#include <hip/hip_runtime.h>

typedef __attribute__((ext_vector_type(16))) __bf16 v16bf;
typedef __attribute__((ext_vector_type(8)))  float  v8f;

__device__ __forceinline__ __bf16 f2bf(float f) {
  union { float f; unsigned u; } in; in.f = f;
  unsigned r = (in.u + 0x7FFFu + ((in.u >> 16) & 1u)) >> 16;
  union { unsigned short s; __bf16 b; } out; out.s = (unsigned short)r;
  return out.b;
}

#define N_NODES 40000
#define N_EDGES 256000

// ---------------- workspace layout (bytes) ----------------
#define OFF_HP   ((size_t)0)                    // 40000*384 f32
#define OFF_HG   ((size_t)61440000)             // 40000*384 f32
#define OFF_AGG  ((size_t)122880000)            // 40000*1152 f32
#define OFF_HGB  ((size_t)307200000)            // 40000*384 bf16
#define OFF_CLSB ((size_t)337920000)            // 40000*5 bf16
#define OFF_WPB  ((size_t)338320000)            // 416*384 bf16 (frag-swizzled)
#define OFF_WGB  ((size_t)338639488)            // 1920*384 bf16 (frag-swizzled)
#define OFF_W1B  ((size_t)340114048)            // 800*304 bf16 (frag-swizzled)
#define OFF_VEC3 ((size_t)340600448)            // 3*384 f32  [bias|gamma|beta] for proj

// ---------------- B-fragment swizzle ----------------
// Output order: frag = (kt*NT + nt); within frag: lane l (0..31), elem e (0..15).
// Element (l,e) of frag (kt,nt) = W[kt*32 + ((l<16)?0:16) + e][nt*16 + (l&15)].
__device__ __forceinline__ void swz_decode(int o, int NT, int& k, int& n) {
  int frag = o >> 9;            // / 512
  int l    = (o >> 4) & 31;
  int e    = o & 15;
  int kt = frag / NT, nt = frag - kt * NT;
  k = kt * 32 + ((l < 16) ? 0 : 16) + e;
  n = nt * 16 + (l & 15);
}

// A-fragment decode for LDS staging: idx -> (row, k-in-tile-range)
// Element (l,e) of tile kt = A[row = l&15][kt*32 + kk],
// kk = akb+e, akb = (l<16)?0:16.
__device__ __forceinline__ void a_decode(int idx, int& kt, int& row, int& kk) {
  kt = idx >> 9;
  int l = (idx >> 4) & 31;
  int e = idx & 15;
  row = l & 15;
  int akb = (l < 16) ? 0 : 16;
  kk = akb + e;
}

// ---------------- small setup kernels ----------------
__global__ void k_vec3(const float* bp0, const float* gp0, const float* bt0,
                       const float* bp1, const float* gp1, const float* bt1,
                       float* vec3) {
  int i = blockIdx.x * blockDim.x + threadIdx.x;
  if (i >= 384) return;
  if (i < 192) { vec3[i] = bp0[i];     vec3[384 + i] = gp0[i];     vec3[768 + i] = bt0[i]; }
  else { int j = i - 192; vec3[i] = bp1[j]; vec3[384 + i] = gp1[j]; vec3[768 + i] = bt1[j]; }
}

// block-diagonal padded proj weight: K=416 (300+pad | 84+pad), N=384 (192|192), frag-swizzled
__global__ void k_wpb(const float* Wp0, const float* Wp1, __bf16* WpB) {
  int idx = blockIdx.x * blockDim.x + threadIdx.x;
  if (idx >= 416 * 384) return;
  int k, n; swz_decode(idx, 24, k, n);
  float v = 0.f;
  if (n < 192) { if (k < 300) v = Wp0[k * 192 + n]; }
  else         { if (k >= 320 && k < 404) v = Wp1[(k - 320) * 192 + (n - 192)]; }
  WpB[idx] = f2bf(v);
}

__global__ void k_wgb(const float* Wg, __bf16* WgB) {
  int idx = blockIdx.x * blockDim.x + threadIdx.x;
  if (idx >= 1920 * 384) return;
  int k, n; swz_decode(idx, 24, k, n);
  WgB[idx] = f2bf(Wg[k * 384 + n]);
}

// W1 padded 784x300 -> 800x304, frag-swizzled
__global__ void k_w1b(const float* W1, __bf16* W1B) {
  int idx = blockIdx.x * blockDim.x + threadIdx.x;
  if (idx >= 800 * 304) return;
  int k, n; swz_decode(idx, 19, k, n);
  float v = (k < 784 && n < 300) ? W1[k * 300 + n] : 0.f;
  W1B[idx] = f2bf(v);
}

__global__ void k_zero(float* p, long n) {
  long i = (long)blockIdx.x * blockDim.x + threadIdx.x;
  long st = (long)gridDim.x * blockDim.x;
  for (; i < n; i += st) p[i] = 0.f;
}

// ---------------- kernel 1: node projection ----------------
__global__ __launch_bounds__(128) void k_proj(const float* __restrict__ h,
                                              const __bf16* __restrict__ WpB,
                                              const float* __restrict__ vec3,
                                              float* __restrict__ hp) {
  __shared__ __align__(32) __bf16 AsF[13 * 512];   // 13 K-tiles, fragment order
  __shared__ float  Cs[16][384];
  __shared__ float  s_mu[2][16], s_rs[2][16];
  const int tid = threadIdx.x;
  const int node0 = blockIdx.x * 16;

  for (int idx = tid; idx < 13 * 512; idx += 128) {
    int kt, row, kk; a_decode(idx, kt, row, kk);
    int c = kt * 32 + kk;
    float v = 0.f;
    if (c < 300)                   v = h[(size_t)(node0 + row) * 384 + c];
    else if (c >= 320 && c < 404)  v = h[(size_t)(node0 + row) * 384 + (c - 20)];
    AsF[idx] = f2bf(v);
  }
  __syncthreads();

  const int wid = tid >> 5, lane = tid & 31;
  const int bn  = lane & 15;
  const int cmb = (lane < 16) ? 0 : 8;
  const v16bf* Af = (const v16bf*)AsF;
  const v16bf* Bf = (const v16bf*)WpB;

  v8f acc[6];
  #pragma unroll
  for (int i = 0; i < 6; ++i) acc[i] = (v8f){0,0,0,0,0,0,0,0};

  for (int kt = 0; kt < 13; ++kt) {
    v16bf a = Af[kt * 32 + lane];
    #pragma unroll
    for (int i = 0; i < 6; ++i) {
      const int nt = wid * 6 + i;
      v16bf b = Bf[(size_t)(kt * 24 + nt) * 32 + lane];
      acc[i] = __builtin_amdgcn_wmma_f32_16x16x32_bf16(false, a, false, b,
                                                       (short)0, acc[i], false, false);
    }
  }
  #pragma unroll
  for (int i = 0; i < 6; ++i) {
    const int nt = wid * 6 + i;
    #pragma unroll
    for (int r = 0; r < 8; ++r) Cs[cmb + r][nt * 16 + bn] = acc[i][r];
  }
  __syncthreads();

  const float* bias = vec3;
  const float* gam  = vec3 + 384;
  const float* bet  = vec3 + 768;
  if (tid < 32) {
    int r = tid & 15, ch = tid >> 4;
    float s = 0.f, s2 = 0.f;
    for (int c = 0; c < 192; ++c) {
      float v = Cs[r][ch * 192 + c] + bias[ch * 192 + c];
      s += v; s2 += v * v;
    }
    float mu = s * (1.f / 192.f);
    float var = s2 * (1.f / 192.f) - mu * mu;
    s_mu[ch][r] = mu;
    s_rs[ch][r] = rsqrtf(var + 1e-5f);
  }
  __syncthreads();
  for (int idx = tid; idx < 16 * 384; idx += 128) {
    int r = idx / 384, c = idx % 384, ch = c / 192;
    float v = Cs[r][c] + bias[c];
    v = (v - s_mu[ch][r]) * s_rs[ch][r] * gam[c] + bet[c];
    hp[(size_t)(node0 + r) * 384 + c] = fmaxf(v, 0.f);
  }
}

// ---------------- kernel 2: edge scatter (segment sum, f32 atomics) ----------------
__global__ __launch_bounds__(128) void k_scatter(const float* __restrict__ hp,
                                                 const int* __restrict__ src,
                                                 const int* __restrict__ dst,
                                                 const int* __restrict__ positions,
                                                 const float* __restrict__ dist,
                                                 float* __restrict__ agg) {
  const int e = blockIdx.x;
  const int pos = positions[e];
  if (pos > 2) return;                 // positions==3 edges contribute nothing
  const float w = dist[e];
  const float* hpr = hp + (size_t)src[e] * 384;
  float* ag = agg + (size_t)dst[e] * 1152 + pos * 384;
  for (int c = threadIdx.x; c < 384; c += 128)
    atomicAdd(&ag[c], hpr[c] * w);
}

// ---------------- kernel 3: hg = relu(LN(hc @ Wg + bg)), hc = [hp, r, b, l, t=r] ----------------
__global__ __launch_bounds__(128) void k_hg(const float* __restrict__ hp,
                                            const float* __restrict__ agg,
                                            const __bf16* __restrict__ WgB,
                                            const float* __restrict__ bg,
                                            const float* __restrict__ gg,
                                            const float* __restrict__ betag,
                                            float* __restrict__ hg,
                                            __bf16* __restrict__ hgB) {
  __shared__ __align__(32) __bf16 AsF[12 * 512];   // one 384-wide K chunk in fragment order
  __shared__ float  Cs[16][384];
  __shared__ float  s_mu[16], s_rs[16];
  const int tid = threadIdx.x;
  const int node0 = blockIdx.x * 16;
  const int wid = tid >> 5, lane = tid & 31;
  const int bn  = lane & 15;
  const int cmb = (lane < 16) ? 0 : 8;
  const v16bf* Af = (const v16bf*)AsF;
  const v16bf* Bf = (const v16bf*)WgB;

  v8f acc[6];
  #pragma unroll
  for (int i = 0; i < 6; ++i) acc[i] = (v8f){0,0,0,0,0,0,0,0};

  for (int chunk = 0; chunk < 5; ++chunk) {
    __syncthreads();
    for (int idx = tid; idx < 12 * 512; idx += 128) {
      int kt, row, kk; a_decode(idx, kt, row, kk);
      int cg = chunk * 384 + kt * 32 + kk;
      float v;
      if (cg < 384)        v = hp[(size_t)(node0 + row) * 384 + cg];
      else if (cg < 1536)  v = agg[(size_t)(node0 + row) * 1152 + (cg - 384)];
      else                 v = agg[(size_t)(node0 + row) * 1152 + (cg - 1536)]; // t = r
      AsF[idx] = f2bf(v);
    }
    __syncthreads();
    for (int kt = 0; kt < 12; ++kt) {
      v16bf a = Af[kt * 32 + lane];
      const int ktg = chunk * 12 + kt;
      #pragma unroll
      for (int i = 0; i < 6; ++i) {
        const int nt = wid * 6 + i;
        v16bf b = Bf[(size_t)(ktg * 24 + nt) * 32 + lane];
        acc[i] = __builtin_amdgcn_wmma_f32_16x16x32_bf16(false, a, false, b,
                                                         (short)0, acc[i], false, false);
      }
    }
  }
  #pragma unroll
  for (int i = 0; i < 6; ++i) {
    const int nt = wid * 6 + i;
    #pragma unroll
    for (int r = 0; r < 8; ++r) Cs[cmb + r][nt * 16 + bn] = acc[i][r];
  }
  __syncthreads();

  if (tid < 16) {
    float s = 0.f, s2 = 0.f;
    for (int c = 0; c < 384; ++c) {
      float v = Cs[tid][c] + bg[c];
      s += v; s2 += v * v;
    }
    float mu = s * (1.f / 384.f);
    float var = s2 * (1.f / 384.f) - mu * mu;
    s_mu[tid] = mu;
    s_rs[tid] = rsqrtf(var + 1e-5f);
  }
  __syncthreads();
  for (int idx = tid; idx < 16 * 384; idx += 128) {
    int r = idx / 384, c = idx % 384;
    float v = Cs[r][c] + bg[c];
    v = (v - s_mu[r]) * s_rs[r] * gg[c] + betag[c];
    v = fmaxf(v, 0.f);
    hg [(size_t)(node0 + r) * 384 + c] = v;
    hgB[(size_t)(node0 + r) * 384 + c] = f2bf(v);
  }
}

// ---------------- kernel 4: node head n = LN(hg@Wn + bn); cls = softmax(n) ----------------
__global__ void k_nhead(const float* __restrict__ hg, const float* __restrict__ Wn,
                        const float* __restrict__ bnv, const float* __restrict__ gn,
                        const float* __restrict__ betan,
                        float* __restrict__ out, __bf16* __restrict__ clsB) {
  int node = blockIdx.x * blockDim.x + threadIdx.x;
  if (node >= N_NODES) return;
  float a[5];
  #pragma unroll
  for (int j = 0; j < 5; ++j) a[j] = bnv[j];
  const float* hr = hg + (size_t)node * 384;
  for (int k = 0; k < 384; ++k) {
    float hv = hr[k];
    #pragma unroll
    for (int j = 0; j < 5; ++j) a[j] += hv * Wn[k * 5 + j];
  }
  float mu = (a[0] + a[1] + a[2] + a[3] + a[4]) * 0.2f;
  float var = 0.f;
  #pragma unroll
  for (int j = 0; j < 5; ++j) { float d = a[j] - mu; var += d * d; }
  var *= 0.2f;
  float rs = rsqrtf(var + 1e-5f);
  float n[5];
  #pragma unroll
  for (int j = 0; j < 5; ++j) {
    n[j] = (a[j] - mu) * rs * gn[j] + betan[j];
    out[(size_t)node * 5 + j] = n[j];
  }
  float mx = n[0];
  #pragma unroll
  for (int j = 1; j < 5; ++j) mx = fmaxf(mx, n[j]);
  float s = 0.f, ex[5];
  #pragma unroll
  for (int j = 0; j < 5; ++j) { ex[j] = __expf(n[j] - mx); s += ex[j]; }
  float inv = 1.f / s;
  #pragma unroll
  for (int j = 0; j < 5; ++j) clsB[(size_t)node * 5 + j] = f2bf(ex[j] * inv);
}

// ---------------- kernel 5: edge MLP ----------------
// x = [hg[src], cls[src], polar, hg[dst], cls[dst]] (784, pad->800)
// e = relu(LN(x@W1 + b1)) @ W2 + b2
__global__ __launch_bounds__(128) void k_edge(const int* __restrict__ src,
                                              const int* __restrict__ dst,
                                              const float* __restrict__ polar,
                                              const __bf16* __restrict__ hgB,
                                              const __bf16* __restrict__ clsB,
                                              const __bf16* __restrict__ W1B,
                                              const float* __restrict__ b1,
                                              const float* __restrict__ g1,
                                              const float* __restrict__ beta1,
                                              const float* __restrict__ W2,
                                              const float* __restrict__ b2,
                                              float* __restrict__ out) {
  __shared__ __align__(32) __bf16 AsF[25 * 512];   // 25 K-tiles, fragment order
  __shared__ float  Cs[16][304];
  __shared__ int    Ss[16], Ds[16];
  __shared__ float  s_mu[16], s_rs[16];
  const int tid = threadIdx.x;
  const int e0 = blockIdx.x * 16;
  if (tid < 16) { Ss[tid] = src[e0 + tid]; Ds[tid] = dst[e0 + tid]; }
  __syncthreads();

  for (int idx = tid; idx < 25 * 512; idx += 128) {
    int kt, row, kk; a_decode(idx, kt, row, kk);
    int c = kt * 32 + kk;
    int s = Ss[row], d = Ds[row];
    __bf16 v;
    if (c < 384)       v = hgB[(size_t)s * 384 + c];
    else if (c < 389)  v = clsB[(size_t)s * 5 + (c - 384)];
    else if (c < 395)  v = f2bf(polar[(size_t)(e0 + row) * 6 + (c - 389)]);
    else if (c < 779)  v = hgB[(size_t)d * 384 + (c - 395)];
    else if (c < 784)  v = clsB[(size_t)d * 5 + (c - 779)];
    else               v = f2bf(0.f);
    AsF[idx] = v;
  }
  __syncthreads();

  const int wid = tid >> 5, lane = tid & 31;
  const int bn  = lane & 15;
  const int cmb = (lane < 16) ? 0 : 8;
  const v16bf* Af = (const v16bf*)AsF;
  const v16bf* Bf = (const v16bf*)W1B;

  v8f acc[5];
  #pragma unroll
  for (int i = 0; i < 5; ++i) acc[i] = (v8f){0,0,0,0,0,0,0,0};

  for (int kt = 0; kt < 25; ++kt) {
    v16bf a = Af[kt * 32 + lane];
    #pragma unroll
    for (int i = 0; i < 5; ++i) {
      const int nt = wid + 4 * i;      // 19 N-tiles over 4 waves
      if (nt >= 19) break;             // wave-uniform
      v16bf b = Bf[(size_t)(kt * 19 + nt) * 32 + lane];
      acc[i] = __builtin_amdgcn_wmma_f32_16x16x32_bf16(false, a, false, b,
                                                       (short)0, acc[i], false, false);
    }
  }
  #pragma unroll
  for (int i = 0; i < 5; ++i) {
    const int nt = wid + 4 * i;
    if (nt >= 19) break;
    #pragma unroll
    for (int r = 0; r < 8; ++r) Cs[cmb + r][nt * 16 + bn] = acc[i][r];
  }
  __syncthreads();

  if (tid < 16) {
    float s = 0.f, s2 = 0.f;
    for (int c = 0; c < 300; ++c) {
      float v = Cs[tid][c] + b1[c];
      s += v; s2 += v * v;
    }
    float mu = s * (1.f / 300.f);
    float var = s2 * (1.f / 300.f) - mu * mu;
    s_mu[tid] = mu;
    s_rs[tid] = rsqrtf(var + 1e-5f);
  }
  __syncthreads();
  for (int idx = tid; idx < 16 * 300; idx += 128) {
    int r = idx / 300, c = idx % 300;
    float v = Cs[r][c] + b1[c];
    v = (v - s_mu[r]) * s_rs[r] * g1[c] + beta1[c];
    Cs[r][c] = fmaxf(v, 0.f);
  }
  __syncthreads();
  if (tid < 32) {
    int r = tid >> 1, cl = tid & 1;
    float s = b2[cl];
    for (int k = 0; k < 300; ++k) s += Cs[r][k] * W2[k * 2 + cl];
    out[(size_t)N_NODES * 5 + (size_t)(e0 + r) * 2 + cl] = s;
  }
}

// ---------------- launch ----------------
extern "C" void kernel_launch(void* const* d_in, const int* in_sizes, int n_in,
                              void* d_out, int out_size, void* d_ws, size_t ws_size,
                              hipStream_t stream) {
  const float* h         = (const float*)d_in[0];
  const int*   src       = (const int*)  d_in[1];
  const int*   dst       = (const int*)  d_in[2];
  const int*   positions = (const int*)  d_in[3];
  const float* dist      = (const float*)d_in[4];
  const float* polar     = (const float*)d_in[5];
  const float* Wp0   = (const float*)d_in[6];
  const float* bp0   = (const float*)d_in[7];
  const float* gp0   = (const float*)d_in[8];
  const float* betap0= (const float*)d_in[9];
  const float* Wp1   = (const float*)d_in[10];
  const float* bp1   = (const float*)d_in[11];
  const float* gp1   = (const float*)d_in[12];
  const float* betap1= (const float*)d_in[13];
  const float* Wg    = (const float*)d_in[14];
  const float* bg    = (const float*)d_in[15];
  const float* gg    = (const float*)d_in[16];
  const float* betag = (const float*)d_in[17];
  const float* Wn    = (const float*)d_in[18];
  const float* bn    = (const float*)d_in[19];
  const float* gn    = (const float*)d_in[20];
  const float* betan = (const float*)d_in[21];
  const float* W1    = (const float*)d_in[22];
  const float* b1    = (const float*)d_in[23];
  const float* g1    = (const float*)d_in[24];
  const float* beta1 = (const float*)d_in[25];
  const float* W2    = (const float*)d_in[26];
  const float* b2    = (const float*)d_in[27];

  char* ws = (char*)d_ws;
  float*  hp   = (float*) (ws + OFF_HP);
  float*  hg   = (float*) (ws + OFF_HG);
  float*  agg  = (float*) (ws + OFF_AGG);
  __bf16* hgB  = (__bf16*)(ws + OFF_HGB);
  __bf16* clsB = (__bf16*)(ws + OFF_CLSB);
  __bf16* WpB  = (__bf16*)(ws + OFF_WPB);
  __bf16* WgB  = (__bf16*)(ws + OFF_WGB);
  __bf16* W1B  = (__bf16*)(ws + OFF_W1B);
  float*  vec3 = (float*) (ws + OFF_VEC3);
  float*  out  = (float*) d_out;

  // weight prep (fragment-swizzled bf16)
  k_vec3<<<3, 128, 0, stream>>>(bp0, gp0, betap0, bp1, gp1, betap1, vec3);
  k_wpb <<<(416 * 384 + 255) / 256, 256, 0, stream>>>(Wp0, Wp1, WpB);
  k_wgb <<<(1920 * 384 + 255) / 256, 256, 0, stream>>>(Wg, WgB);
  k_w1b <<<(800 * 304 + 255) / 256, 256, 0, stream>>>(W1, W1B);

  // node projection (WMMA)
  k_proj<<<N_NODES / 16, 128, 0, stream>>>(h, WpB, vec3, hp);

  // scatter (segment sums)
  k_zero<<<8192, 256, 0, stream>>>(agg, (long)N_NODES * 1152);
  k_scatter<<<N_EDGES, 128, 0, stream>>>(hp, src, dst, positions, dist, agg);

  // hidden GEMM (WMMA)
  k_hg<<<N_NODES / 16, 128, 0, stream>>>(hp, agg, WgB, bg, gg, betag, hg, hgB);

  // node head -> n (output) + cls (bf16)
  k_nhead<<<(N_NODES + 127) / 128, 128, 0, stream>>>(hg, Wn, bn, gn, betan, out, clsB);

  // edge MLP (WMMA) -> e (output)
  k_edge<<<N_EDGES / 16, 128, 0, stream>>>(src, dst, polar, hgB, clsB,
                                           W1B, b1, g1, beta1, W2, b2, out);
}
